// SpatialEmbLoss_88965952569690
// MI455X (gfx1250) — matrix operand order repaired and checked by reading, not compiled, not run on hardware
//
#include <hip/hip_runtime.h>
#include <hip/hip_bf16.h>
#include <math.h>

typedef __attribute__((ext_vector_type(16))) _Float16 v16h;
typedef __attribute__((ext_vector_type(8)))  float    v8f;

#define HH 512
#define WWI 512
#define HWPIX (HH*WWI)
#define NB 1024               // lovasz histogram bins over err in [0,2]
#define ACC_STRIDE 256        // floats per image in accumulator region
// per-image accumulator layout (float offsets)
#define PSTAT 0               // 16 instances x 5 stats [cnt, sx, sy, ss, ss2]
#define OSTAT 80              // 8 instances x 5
#define PMETA 120             // 16 x 3 [cx, cy, l11]
#define OMETA 168             // 8 x 3
#define SCAL  192             // [p_bg, o_bg, p_seed, o_seed, p_var, o_var, p_inst, o_inst]
#define HIST_OFF_U32 1024     // histograms start at byte 4096 of ws

__device__ __forceinline__ float sigmoidf_(float x) {
    return 1.f / (1.f + __expf(-x));
}

__global__ void k_zero(float* acc) {
    int i = blockIdx.x * blockDim.x + threadIdx.x;
    if (i < 4 * ACC_STRIDE) acc[i] = 0.f;
}

// ---------------------------------------------------------------------------
// Pass 1: per-instance segmented reductions as WMMA mask-GEMM.
//   A[16x32] f16 : one-hot instance masks for 32 pixels
//   B[32x16] f16 : columns [1, x, y, sigma, sigma^2, 0...] per pixel
//   C[16x16] f32 : accumulates per-instance stats (f32 accumulate).
// B-fragment build is branchless: per-wave LDS stat table with 8 rows
//   [1, x, y, sp, sp^2, so, so^2, 0]; each lane's row index is fixed.
// Also accumulates background seed losses.
// ---------------------------------------------------------------------------
__global__ void k_stats(const float* __restrict__ pred,
                        const int*  __restrict__ obj_inst,
                        const int*  __restrict__ obj_lab,
                        const int*  __restrict__ parts_inst,
                        const int*  __restrict__ parts_lab,
                        const float* __restrict__ xym,
                        float* __restrict__ acc) {
    const int b    = blockIdx.y;
    const int tid  = threadIdx.x;
    const int wave = tid >> 5;
    const int lane = tid & 31;
    const bool hi  = lane >= 16;
    const int  mn  = lane & 15;           // M for A-frag, N for B/C-frags

    __shared__ float sT[8][8][32];        // [wave][row][pixel] stat table
    __shared__ int   sPid[8][32], sOid[8][32];

    // row selection per column N (uniform per lane across all iterations)
    // parts: cols {1,x,y,sp,sp2}; obj: cols {1,x,y,so,so2}; others -> zero row
    const int rowP = (mn == 0) ? 0 : (mn == 1) ? 1 : (mn == 2) ? 2
                   : (mn == 3) ? 3 : (mn == 4) ? 4 : 7;
    const int rowO = (mn == 0) ? 0 : (mn == 1) ? 1 : (mn == 2) ? 2
                   : (mn == 3) ? 5 : (mn == 4) ? 6 : 7;

    // constant rows written once (visible after first in-loop barrier)
    sT[wave][0][lane] = 1.f;
    sT[wave][7][lane] = 0.f;

    const float* predb = pred + (size_t)b * 8 * HWPIX;
    const int*   oi = obj_inst   + (size_t)b * HWPIX;
    const int*   pi = parts_inst + (size_t)b * HWPIX;
    const int*   ol = obj_lab    + (size_t)b * HWPIX;
    const int*   pl = parts_lab  + (size_t)b * HWPIX;

    v8f accP = {};
    v8f accO = {};
    float bgO = 0.f, bgP = 0.f;

    const int gwave  = blockIdx.x * 8 + wave;   // 0..511  (gridDim.x == 64)
    const int nchunk = HWPIX / 32;              // 8192 -> 16 uniform iters/wave

    for (int c = gwave; c < nchunk; c += 512) {
        const int p   = c * 32 + lane;
        const int pid = pi[p];
        const int oid = oi[p];
        const float xm = xym[p];
        const float ym = xym[HWPIX + p];
        const float sp = predb[5 * HWPIX + p];  // parts sigma
        const float so = predb[2 * HWPIX + p];  // obj sigma
        const float osd = sigmoidf_(predb[6 * HWPIX + p]);
        const float psd = sigmoidf_(predb[7 * HWPIX + p]);
        if (ol[p] != 1) bgO += osd * osd;       // background seed regression
        if (pl[p] != 1) bgP += psd * psd;

        __syncthreads();                        // WAR: prev iter's reads done
        sT[wave][1][lane] = xm;
        sT[wave][2][lane] = ym;
        sT[wave][3][lane] = sp;
        sT[wave][4][lane] = sp * sp;
        sT[wave][5][lane] = so;
        sT[wave][6][lane] = so * so;
        sPid[wave][lane] = pid;
        sOid[wave][lane] = oid;
        __syncthreads();

        v16h aP, aO, bP, bO;
#pragma unroll
        for (int e = 0; e < 16; ++e) {
            // A layout (16-bit 16x32): lane<16 K in {0..7,16..23}, lane>=16 +8
            const int ka = e + ((e >= 8) ? 8 : 0) + (hi ? 8 : 0);
            aP[e] = (_Float16)((sPid[wave][ka] == mn + 1) ? 1.f : 0.f);
            aO[e] = (_Float16)((sOid[wave][ka] == mn + 1) ? 1.f : 0.f);
            // B layout (32x16): same K order as A (lane<16 K in {0..7,16..23}, lane>=16 +8), N=lane&15
            const int kb = ka;
            bP[e] = (_Float16)sT[wave][rowP][kb];   // branchless row gather
            bO[e] = (_Float16)sT[wave][rowO][kb];
        }
        accP = __builtin_amdgcn_wmma_f32_16x16x32_f16(false, aP, false, bP,
                                                      (short)0, accP, false, false);
        accO = __builtin_amdgcn_wmma_f32_16x16x32_f16(false, aO, false, bO,
                                                      (short)0, accO, false, false);
    }

    // C/D layout: VGPR j -> M = j + (hi?8:0), N = lane&15
    float* A = acc + b * ACC_STRIDE;
    if (mn < 5) {
#pragma unroll
        for (int j = 0; j < 8; ++j) {
            const int m = j + (hi ? 8 : 0);
            atomicAdd(&A[PSTAT + m * 5 + mn], accP[j]);
        }
        if (!hi) {
#pragma unroll
            for (int j = 0; j < 8; ++j)
                atomicAdd(&A[OSTAT + j * 5 + mn], accO[j]);
        }
    }
    // wave-reduce bg seed terms (wave32)
#pragma unroll
    for (int o = 16; o; o >>= 1) {
        bgP += __shfl_down(bgP, o, 32);
        bgO += __shfl_down(bgO, o, 32);
    }
    if (lane == 0) {
        atomicAdd(&A[SCAL + 0], bgP);
        atomicAdd(&A[SCAL + 1], bgO);
    }
}

// ---------------------------------------------------------------------------
// Pass 2: finalize per-instance meta: centroid, l11 = exp(5.5*s), var term.
// ---------------------------------------------------------------------------
__global__ void k_finalize(float* acc) {
    const int t = threadIdx.x;
    if (t >= 96) return;
    const int b = t / 24, i = t % 24;
    float* A = acc + b * ACC_STRIDE;
    const float* st = (i < 16) ? &A[PSTAT + i * 5] : &A[OSTAT + (i - 16) * 5];
    float*       mt = (i < 16) ? &A[PMETA + i * 3] : &A[OMETA + (i - 16) * 3];
    const float cnt = st[0];
    float cx = 0.f, cy = 0.f, l11 = 0.f;
    if (cnt > 0.f) {
        cx = st[1] / cnt;
        cy = st[2] / cnt;
        const float s = st[3] / cnt;
        const float var = st[4] / cnt - s * s;   // Sum(mask*(sig - s)^2)/cnt
        l11 = __expf(5.5f * s);
        atomicAdd(&A[SCAL + ((i < 16) ? 4 : 5)], var);
    }
    mt[0] = cx; mt[1] = cy; mt[2] = l11;
}

// ---------------------------------------------------------------------------
// Pass 3: one workgroup per (image, instance). Compute prob map, masked seed
// loss, and a 1024-bin LDS histogram of lovasz errors (err in [0,2]) with
// per-bin (count, gt-count, sum_err, sum_err_gt). Flush to global.
// ---------------------------------------------------------------------------
__global__ void k_hist(const float* __restrict__ pred,
                       const int*  __restrict__ obj_inst,
                       const int*  __restrict__ parts_inst,
                       const float* __restrict__ xym,
                       float* __restrict__ acc,
                       unsigned* __restrict__ ws) {
    const int slot = blockIdx.x;          // 0..95
    const int b    = slot / 24;
    const int idx  = slot % 24;
    const bool isPart = idx < 16;
    const int n = isPart ? idx : idx - 16;

    float* A = acc + b * ACC_STRIDE;
    const float* mt = isPart ? &A[PMETA + n * 3] : &A[OMETA + n * 3];
    const float cx = mt[0], cy = mt[1], l11 = mt[2];

    __shared__ unsigned hC[NB], hG[NB];
    __shared__ float hE[NB], hEg[NB];
    for (int i = threadIdx.x; i < NB; i += blockDim.x) {
        hC[i] = 0u; hG[i] = 0u; hE[i] = 0.f; hEg[i] = 0.f;
    }
    __syncthreads();

    const float* predb = pred + (size_t)b * 8 * HWPIX;
    const int* ids = (isPart ? parts_inst : obj_inst) + (size_t)b * HWPIX;
    float seedAcc = 0.f;

    for (int p = threadIdx.x; p < HWPIX; p += blockDim.x) {
        const int gt = (ids[p] == n + 1) ? 1 : 0;
        const float xm = xym[p], ym = xym[HWPIX + p];
        float ex = xm + tanhf(predb[3 * HWPIX + p]);   // parts_emb
        float ey = ym + tanhf(predb[4 * HWPIX + p]);
        if (!isPart) {                                  // obj_emb = parts_emb + tanh(pred[0:2])
            ex += tanhf(predb[0 * HWPIX + p]);
            ey += tanhf(predb[1 * HWPIX + p]);
        }
        const float dx = ex - cx, dy = ey - cy;
        const float prob = __expf(-0.5f * l11 * (dx * dx + dy * dy));
        if (gt) {
            const float seed = sigmoidf_(predb[(isPart ? 7 : 6) * HWPIX + p]);
            const float d = seed - prob;
            seedAcc += d * d;                           // CLASS_W == 1
        }
        const float err = gt ? (2.f - 2.f * prob) : (2.f * prob);
        int bin = (int)(err * (NB * 0.5f));
        bin = bin < 0 ? 0 : (bin > NB - 1 ? NB - 1 : bin);
        atomicAdd(&hC[bin], 1u);
        atomicAdd(&hE[bin], err);
        if (gt) { atomicAdd(&hG[bin], 1u); atomicAdd(&hEg[bin], err); }
    }
    __syncthreads();

    unsigned* hb = ws + HIST_OFF_U32 + (size_t)slot * 4 * NB;
    for (int i = threadIdx.x; i < NB; i += blockDim.x) {
        hb[i]          = hC[i];
        hb[NB + i]     = hG[i];
        ((float*)hb)[2 * NB + i] = hE[i];
        ((float*)hb)[3 * NB + i] = hEg[i];
    }

    __shared__ float red[256];
    red[threadIdx.x] = seedAcc;
    __syncthreads();
    for (int o = 128; o; o >>= 1) {
        if (threadIdx.x < o) red[threadIdx.x] += red[threadIdx.x + o];
        __syncthreads();
    }
    if (threadIdx.x == 0) atomicAdd(&A[SCAL + (isPart ? 2 : 3)], red[0]);
}

// ---------------------------------------------------------------------------
// Pass 4: lovasz hinge from histogram. Descending-error order == descending
// bins. Exclusive prefix scan of (gt, non-gt) counts, then telescoped
// jaccard deltas per bin: sum(es*delta) ~= mean_e * (jac_end - jac_start).
// ---------------------------------------------------------------------------
__global__ void k_lovasz(float* __restrict__ acc, const unsigned* __restrict__ ws) {
    const int slot = blockIdx.x;
    const int b    = slot / 24;
    const int idx  = slot % 24;
    const bool isPart = idx < 16;

    const unsigned* hb  = ws + HIST_OFF_U32 + (size_t)slot * 4 * NB;
    const unsigned* hC  = hb;
    const unsigned* hG  = hb + NB;
    const float*    hE  = (const float*)(hb + 2 * NB);
    const float*    hEg = (const float*)(hb + 3 * NB);

    const int t = threadIdx.x;            // 256 threads
    const int PER = NB / 256;             // 4 bins/thread

    float lg = 0.f, ln = 0.f;
#pragma unroll
    for (int j = 0; j < PER; ++j) {
        const int bin = NB - 1 - (t * PER + j);
        const float c1 = (float)hG[bin];
        lg += c1;
        ln += (float)hC[bin] - c1;
    }
    __shared__ float sG[256], sN[256];
    sG[t] = lg; sN[t] = ln;
    __syncthreads();
    for (int o = 1; o < 256; o <<= 1) {   // Hillis-Steele inclusive scan
        const float vg = (t >= o) ? sG[t - o] : 0.f;
        const float vn = (t >= o) ? sN[t - o] : 0.f;
        __syncthreads();
        sG[t] += vg; sN[t] += vn;
        __syncthreads();
    }
    const float G = sG[255];              // total gt pixels
    float CG = sG[t] - lg;                // exclusive prefixes
    float CN = sN[t] - ln;

    float contrib = 0.f;
    if (G > 0.f) {
        for (int j = 0; j < PER; ++j) {
            const int bin = NB - 1 - (t * PER + j);
            const float c1 = (float)hG[bin];
            const float cA = (float)hC[bin];
            const float c0 = cA - c1;
            const float e1 = hEg[bin];
            const float e0 = hE[bin] - e1;
            if (c1 > 0.f) {               // gt=1 run: inter shrinks
                const float j0 = 1.f - (G - CG) / (G + CN);
                CG += c1;
                const float j1 = 1.f - (G - CG) / (G + CN);
                contrib += (e1 / c1) * (j1 - j0);
            }
            if (c0 > 0.f) {               // gt=0 run: union grows
                const float j0 = 1.f - (G - CG) / (G + CN);
                CN += c0;
                const float j1 = 1.f - (G - CG) / (G + CN);
                contrib += (e0 / c0) * (j1 - j0);
            }
        }
    }
    __shared__ float red[256];
    red[t] = contrib;
    __syncthreads();
    for (int o = 128; o; o >>= 1) {
        if (t < o) red[t] += red[t + o];
        __syncthreads();
    }
    if (t == 0) atomicAdd(&acc[b * ACC_STRIDE + SCAL + (isPart ? 6 : 7)], red[0]);
}

// ---------------------------------------------------------------------------
// Pass 5: combine per-image terms. W_INST=1, W_VAR=10, W_SEED=1, W_OFFSET=0.
// ---------------------------------------------------------------------------
__global__ void k_combine(const float* __restrict__ acc, float* __restrict__ out) {
    if (threadIdx.x != 0 || blockIdx.x != 0) return;
    float total = 0.f;
    const float npix = (float)HWPIX;
    for (int b = 0; b < 4; ++b) {
        const float* A = acc + b * ACC_STRIDE;
        const float p_bg = A[SCAL + 0], o_bg = A[SCAL + 1];
        const float p_sd = A[SCAL + 2], o_sd = A[SCAL + 3];
        const float p_vr = A[SCAL + 4], o_vr = A[SCAL + 5];
        const float p_in = A[SCAL + 6], o_in = A[SCAL + 7];
        total += (o_in / 8.f + p_in / 16.f)
               + 10.f * (o_vr / 8.f + p_vr / 16.f)
               + ((o_bg + o_sd) / npix + (p_bg + p_sd) / npix);
    }
    out[0] = total / 4.f;
}

extern "C" void kernel_launch(void* const* d_in, const int* in_sizes, int n_in,
                              void* d_out, int out_size, void* d_ws, size_t ws_size,
                              hipStream_t stream) {
    const float* pred       = (const float*)d_in[0];   // [4,8,512,512] f32
    const int*   obj_inst   = (const int*)d_in[1];     // [4,512,512] i32
    const int*   obj_lab    = (const int*)d_in[2];
    const int*   parts_inst = (const int*)d_in[3];
    const int*   parts_lab  = (const int*)d_in[4];
    // d_in[5] stem_anno unused
    const float* xym        = (const float*)d_in[6];   // [2,512,512] f32
    float* out = (float*)d_out;

    float*    acc = (float*)d_ws;
    unsigned* ws  = (unsigned*)d_ws;

    k_zero<<<dim3(4), dim3(256), 0, stream>>>(acc);
    k_stats<<<dim3(64, 4), dim3(256), 0, stream>>>(pred, obj_inst, obj_lab,
                                                   parts_inst, parts_lab, xym, acc);
    k_finalize<<<dim3(1), dim3(128), 0, stream>>>(acc);
    k_hist<<<dim3(96), dim3(256), 0, stream>>>(pred, obj_inst, parts_inst, xym, acc, ws);
    k_lovasz<<<dim3(96), dim3(256), 0, stream>>>(acc, ws);
    k_combine<<<dim3(1), dim3(32), 0, stream>>>(acc, out);
}
